// Attention_87557203296260
// MI455X (gfx1250) — matrix operand order repaired and checked by reading, not compiled, not run on hardware
//
#include <hip/hip_runtime.h>
#include <hip/hip_bf16.h>
#include <cstdint>
#include <cstddef>
#include <cmath>

// ---------------------------------------------------------------------------
// Types for gfx1250 WMMA (wave32): A/B = v16h (8 VGPRs of f16 pairs),
// C/D = v8f (8 VGPRs f32).
// ---------------------------------------------------------------------------
typedef __attribute__((ext_vector_type(16))) _Float16 v16h;
typedef __attribute__((ext_vector_type(8)))  _Float16 v8h;
typedef __attribute__((ext_vector_type(8)))  float    v8f;

#define WMMA_F16(a, b, c) \
  __builtin_amdgcn_wmma_f32_16x16x32_f16(false, (a), false, (b), (short)0, (c), false, false)

// Raw hardware exp2 (v_exp_f32).
static __device__ __forceinline__ float hw_exp2(float x) {
#if __has_builtin(__builtin_amdgcn_exp2f)
  return __builtin_amdgcn_exp2f(x);
#else
  return exp2f(x);
#endif
}

// A-matrix 16x32 f16 layout: lane -> M (lane&15); halves h -> K = h + 16*hi
// (hi = lane>=16), matching the B operand order.
// rowp points at (row m, k=0) of a row-major source.
static __device__ __forceinline__ v16h ldA(const _Float16* rowp, int hi) {
  return *(const v16h*)(rowp + 16 * hi);
}

// B-matrix 32x16 f16 layout: lane -> N (lane&15); halves h -> K = h + 16*hi.
// colp points at (k=0) of this lane's column, contiguous over k.
static __device__ __forceinline__ v16h ldB(const _Float16* colp, int hi) {
  return *(const v16h*)(colp + 16 * hi);
}

// ---------------------------------------------------------------------------
// Lane reductions over 16-lane halves.  Prefer v_permlane16_b32 (pure VALU
// crossbar, keeps the LDS pipe free for the P-tile transpose) over
// ds_bpermute-based __shfl_xor.  Selector nibbles: sel[i] = i ^ MASK.
// ---------------------------------------------------------------------------
template <int MASK, unsigned LO, unsigned HI>
static __device__ __forceinline__ float xor16(float v) {
#if __has_builtin(__builtin_amdgcn_permlane16)
  int i = __float_as_int(v);
  return __int_as_float(__builtin_amdgcn_permlane16(i, i, (int)LO, (int)HI, false, false));
#else
  return __shfl_xor(v, MASK);
#endif
}

static __device__ __forceinline__ float red_max16(float v) {
  v = fmaxf(v, xor16<1, 0x67452301u, 0xEFCDAB89u>(v));
  v = fmaxf(v, xor16<2, 0x54761032u, 0xDCFE98BAu>(v));
  v = fmaxf(v, xor16<4, 0x32107654u, 0xBA98FEDCu>(v));
  v = fmaxf(v, xor16<8, 0xFEDCBA98u, 0x76543210u>(v));
  return v;
}

static __device__ __forceinline__ float red_sum16(float v) {
  v += xor16<1, 0x67452301u, 0xEFCDAB89u>(v);
  v += xor16<2, 0x54761032u, 0xDCFE98BAu>(v);
  v += xor16<4, 0x32107654u, 0xBA98FEDCu>(v);
  v += xor16<8, 0xFEDCBA98u, 0x76543210u>(v);
  return v;
}

// ---------------------------------------------------------------------------
// Problem constants (from the reference)
// ---------------------------------------------------------------------------
#define BB    4
#define NN    2048
#define DIM_  1024
#define HH    16
#define DHD   64
#define MROWS (BB * NN)      /* 8192 */
#define N3    (3 * DIM_)     /* 3072 */

// ---------------------------------------------------------------------------
// Kernel 1: fp32 -> f16 elementwise
// ---------------------------------------------------------------------------
__global__ void f32_to_f16(const float* __restrict__ src, _Float16* __restrict__ dst, int n) {
  int i = blockIdx.x * blockDim.x + threadIdx.x;
  if (i < n) dst[i] = (_Float16)src[i];
}

// ---------------------------------------------------------------------------
// Kernel 2: pre-pack a row-major fp32 weight [K x N] into WMMA B-tile layout.
// Packed index: ((tk * (N/16) + tn) * 32 + lane) * 16 + h  with
// k = tk*32 + h + 16*(lane>>4), n = tn*16 + (lane&15).
// The GEMM then loads one B tile as a single contiguous v16h per lane.
// ---------------------------------------------------------------------------
__global__ void pack_b(const float* __restrict__ w, _Float16* __restrict__ dst, int K, int N) {
  int i = blockIdx.x * blockDim.x + threadIdx.x;
  if (i >= K * N) return;
  int tile = i >> 9;
  int within = i & 511;
  int lane = within >> 4;
  int h = within & 15;
  int hi = lane >> 4;
  int nl = lane & 15;
  int tiles_n = N >> 4;
  int tk = tile / tiles_n;
  int tn = tile - tk * tiles_n;
  int k = tk * 32 + h + 16 * hi;
  int n = tn * 16 + nl;
  dst[i] = (_Float16)w[(size_t)k * N + n];
}

// ---------------------------------------------------------------------------
// Kernel 3/6: f16 WMMA GEMM.  C[M,N] = A[M,K] * B[K,N] (+ bias).
// Wave tile = 64x64 (4x4 accumulator tiles, 16 WMMAs per 32-k step,
// 16 b128-class loads per step -> 1 load per WMMA).
// Block = 128 threads = 4 waves arranged 2x2 -> 128x128 block tile.
// ---------------------------------------------------------------------------
template <bool OUT_F16, bool BIAS>
__global__ void __launch_bounds__(128)
gemm_wmma(const _Float16* __restrict__ A, const _Float16* __restrict__ Bp,
          void* __restrict__ Cout, const float* __restrict__ bias,
          int M, int N, int K) {
  const int lane = threadIdx.x & 31;
  const int wave = threadIdx.x >> 5;
  const int ml = lane & 15;
  const int hi = lane >> 4;
  const int nblk = N >> 7;
  const int bm = blockIdx.x / nblk;
  const int bn = blockIdx.x - bm * nblk;
  const int m0 = bm * 128 + (wave >> 1) * 64;
  const int n0 = bn * 128 + (wave & 1) * 64;
  const int tiles_n = N >> 4;

  v8f acc[4][4];
#pragma unroll
  for (int i = 0; i < 4; ++i)
#pragma unroll
    for (int j = 0; j < 4; ++j) acc[i][j] = {};

  const _Float16* ap[4];
#pragma unroll
  for (int i = 0; i < 4; ++i) ap[i] = A + (size_t)(m0 + 16 * i + ml) * K;
  const _Float16* bbase = Bp + (size_t)(n0 >> 4) * 512 + lane * 16;

  for (int kt = 0; kt < K; kt += 32) {
    v16h a[4];
#pragma unroll
    for (int i = 0; i < 4; ++i) a[i] = ldA(ap[i] + kt, hi);

    const _Float16* bp = bbase + (size_t)(kt >> 5) * tiles_n * 512;
    v16h b0 = *(const v16h*)(bp);
    v16h b1 = *(const v16h*)(bp + 512);
    v16h b2 = *(const v16h*)(bp + 1024);
    v16h b3 = *(const v16h*)(bp + 1536);

#pragma unroll
    for (int i = 0; i < 4; ++i) {
      acc[i][0] = WMMA_F16(a[i], b0, acc[i][0]);
      acc[i][1] = WMMA_F16(a[i], b1, acc[i][1]);
      acc[i][2] = WMMA_F16(a[i], b2, acc[i][2]);
      acc[i][3] = WMMA_F16(a[i], b3, acc[i][3]);
    }
  }

  // C/D layout: VGPR r, lanes 0-15 -> M = r, lanes 16-31 -> M = r+8; lane&15 -> N.
#pragma unroll
  for (int i = 0; i < 4; ++i) {
    const int rowb = m0 + i * 16 + 8 * hi;
#pragma unroll
    for (int j = 0; j < 4; ++j) {
      const int col = n0 + j * 16 + ml;
      const float bv = BIAS ? bias[col] : 0.0f;
#pragma unroll
      for (int r = 0; r < 8; ++r) {
        float v = acc[i][j][r] + bv;
        if (OUT_F16)
          ((_Float16*)Cout)[(size_t)(rowb + r) * N + col] = (_Float16)v;
        else
          ((float*)Cout)[(size_t)(rowb + r) * N + col] = v;
      }
    }
  }
}

// ---------------------------------------------------------------------------
// Kernel 4: per-head V transpose: vt[(b*H+h)*64 + d][2048 keys] = V[b,key,h,d]
// so PV B-tiles are contiguous over the key (k) dimension.
// ---------------------------------------------------------------------------
__global__ void v_transpose(const _Float16* __restrict__ qkv, _Float16* __restrict__ vt) {
  int i = blockIdx.x * blockDim.x + threadIdx.x;  // B*H*DH*N = 8,388,608
  int key = i & (NN - 1);
  int d = (i >> 11) & (DHD - 1);
  int bh = i >> 17;
  int b = bh >> 4;
  int h = bh & 15;
  vt[i] = qkv[(size_t)(b * NN + key) * N3 + 2 * DIM_ + h * DHD + d];
}

// ---------------------------------------------------------------------------
// Kernel 5: flash attention. One wave owns 16 query rows of one (b,h).
// Keys in chunks of 64: 8 WMMAs for S = Q*K^T, online softmax in exp2 domain
// (scale * log2(e) pre-folded into the Q register tile; row reductions via
// v_permlane16 VALU crossbar), P tile C-layout -> A-layout through a 2 KB
// per-wave LDS slab, 8 WMMAs for O += P*V.  Next chunk's K/V prefetched.
// ---------------------------------------------------------------------------
__global__ void __launch_bounds__(128)
attn_flash(const _Float16* __restrict__ qkv, const _Float16* __restrict__ vt,
           const unsigned char* __restrict__ mask, _Float16* __restrict__ attn) {
  __shared__ _Float16 lds[4][16 * 64];
  const int lane = threadIdx.x & 31;
  const int wave = threadIdx.x >> 5;
  const int ml = lane & 15;
  const int hi = lane >> 4;
  const int gw = blockIdx.x * 4 + wave;
  const int qt = gw & ((NN / 16) - 1);  // query tile 0..127
  const int bh = gw >> 7;               // 0..63
  const int b = bh >> 4;
  const int h = bh & 15;
  const int q0 = qt << 4;

  // scale * log2(e): softmax runs in the exp2 domain -> raw v_exp_f32.
  const float QSCALE = 0.125f * 1.44269504088896340736f;
  const float NEG = -3.4028234e38f;  // matches -finfo(f32).max

  // Q tile (held in registers for the whole key loop), pre-scaled.
  const _Float16* qrow = qkv + (size_t)(b * NN + q0 + ml) * N3 + h * DHD;
  v16h qa0 = ldA(qrow, hi);        // d = 0..31
  v16h qa1 = ldA(qrow + 32, hi);   // d = 32..63
#pragma unroll
  for (int z = 0; z < 16; ++z) {
    qa0[z] = (_Float16)((float)qa0[z] * QSCALE);
    qa1[z] = (_Float16)((float)qa1[z] * QSCALE);
  }

  v8f o[4];
#pragma unroll
  for (int j = 0; j < 4; ++j) o[j] = {};
  float rm[8], rs[8];
#pragma unroll
  for (int r = 0; r < 8; ++r) { rm[r] = NEG; rs[r] = 0.0f; }

  const _Float16* kbase = qkv + (size_t)b * NN * N3 + DIM_ + h * DHD;
  const _Float16* vbase = vt + (size_t)bh * DHD * NN;
  const unsigned char* mrow = mask + b * NN;
  _Float16* pl = lds[wave];

  for (int k0 = 0; k0 < NN; k0 += 64) {
    // ---- S = Q * K^T over 4 key sub-tiles (keys k0 + 16*t + ml) ----
    v8f s[4];
#pragma unroll
    for (int t = 0; t < 4; ++t) {
      const _Float16* kr = kbase + (size_t)(k0 + 16 * t + ml) * N3;
      v8f z = {};
      z = WMMA_F16(qa0, ldB(kr, hi), z);
      z = WMMA_F16(qa1, ldB(kr + 32, hi), z);
      s[t] = z;
    }

    // Prefetch next chunk's K rows and next chunk's V rows into cache.
    if (k0 + 64 < NN) {
#pragma unroll
      for (int t = 0; t < 4; ++t)
        __builtin_prefetch(kbase + (size_t)(k0 + 64 + 16 * t + ml) * N3, 0, 3);
#pragma unroll
      for (int j = 0; j < 4; ++j)
        __builtin_prefetch(vbase + (size_t)(j * 16 + ml) * NN + k0 + 64, 0, 3);
    }

    bool mk[4];
#pragma unroll
    for (int t = 0; t < 4; ++t) mk[t] = mrow[k0 + 16 * t + ml] != 0;

    // ---- online softmax (exp2 domain) over the 64-key chunk ----
#pragma unroll
    for (int r = 0; r < 8; ++r) {
      float t0 = mk[0] ? s[0][r] : NEG;
      float t1 = mk[1] ? s[1][r] : NEG;
      float t2 = mk[2] ? s[2][r] : NEG;
      float t3 = mk[3] ? s[3][r] : NEG;
      // row max across the 16 columns (lanes) of this half
      float cm = red_max16(fmaxf(fmaxf(t0, t1), fmaxf(t2, t3)));
      float nm = fmaxf(rm[r], cm);
      float corr = hw_exp2(rm[r] - nm);
      rm[r] = nm;
      t0 = hw_exp2(t0 - nm);
      t1 = hw_exp2(t1 - nm);
      t2 = hw_exp2(t2 - nm);
      t3 = hw_exp2(t3 - nm);
      float cs = red_sum16((t0 + t1) + (t2 + t3));
      rs[r] = rs[r] * corr + cs;
#pragma unroll
      for (int j = 0; j < 4; ++j) o[j][r] *= corr;
      // P tile -> LDS (C-layout scatter, row-major [16][64] f16)
      const int rowoff = (r + 8 * hi) * 64;
      pl[rowoff + ml] = (_Float16)t0;
      pl[rowoff + 16 + ml] = (_Float16)t1;
      pl[rowoff + 32 + ml] = (_Float16)t2;
      pl[rowoff + 48 + ml] = (_Float16)t3;
    }

    // LDS ops from one wave execute in order; reload P in A-layout (16x64
    // -> two 16x32 A tiles).
    v16h pa0 = ldA(pl + ml * 64, hi);
    v16h pa1 = ldA(pl + ml * 64 + 32, hi);

    // ---- O += P * V ----
#pragma unroll
    for (int j = 0; j < 4; ++j) {
      const _Float16* vr = vbase + (size_t)(j * 16 + ml) * NN + k0;
      o[j] = WMMA_F16(pa0, ldB(vr, hi), o[j]);
      o[j] = WMMA_F16(pa1, ldB(vr + 32, hi), o[j]);
    }
  }

  // Normalize and store to attn buffer [8192 x 1024] f16 (row = b*N+q, col = h*64+d)
  _Float16* orow = attn + (size_t)(b * NN + q0) * DIM_ + h * DHD;
#pragma unroll
  for (int r = 0; r < 8; ++r) {
    const float inv = rs[r] > 0.0f ? 1.0f / rs[r] : 0.0f;
    const int row = r + 8 * hi;
#pragma unroll
    for (int j = 0; j < 4; ++j) {
      orow[(size_t)row * DIM_ + j * 16 + ml] = (_Float16)(o[j][r] * inv);
    }
  }
}

// ---------------------------------------------------------------------------
// Host-side launch
// ---------------------------------------------------------------------------
extern "C" void kernel_launch(void* const* d_in, const int* in_sizes, int n_in,
                              void* d_out, int out_size, void* d_ws, size_t ws_size,
                              hipStream_t stream) {
  (void)in_sizes; (void)n_in; (void)out_size; (void)ws_size;

  const float* x = (const float*)d_in[0];
  const unsigned char* mask = (const unsigned char*)d_in[1];
  const float* w_qkv = (const float*)d_in[2];
  const float* w_out = (const float*)d_in[3];
  const float* b_out = (const float*)d_in[4];

  char* ws = (char*)d_ws;
  size_t off = 0;
  _Float16* xh = (_Float16*)(ws + off);    off += (size_t)MROWS * DIM_ * 2;   // 16.8 MB
  _Float16* wqkvp = (_Float16*)(ws + off); off += (size_t)DIM_ * N3 * 2;      //  6.3 MB
  _Float16* woutp = (_Float16*)(ws + off); off += (size_t)DIM_ * DIM_ * 2;    //  2.1 MB
  _Float16* qkvh = (_Float16*)(ws + off);  off += (size_t)MROWS * N3 * 2;     // 50.3 MB
  _Float16* vth = (_Float16*)(ws + off);   off += (size_t)BB * HH * DHD * NN * 2; // 16.8 MB
  _Float16* attnh = (_Float16*)(ws + off);                                    // 16.8 MB

  // 1) x -> f16
  f32_to_f16<<<(MROWS * DIM_) / 256, 256, 0, stream>>>(x, xh, MROWS * DIM_);
  // 2) pre-pack weights into WMMA B-tile layout
  pack_b<<<(DIM_ * N3) / 256, 256, 0, stream>>>(w_qkv, wqkvp, DIM_, N3);
  pack_b<<<(DIM_ * DIM_) / 256, 256, 0, stream>>>(w_out, woutp, DIM_, DIM_);
  // 3) qkv = x @ w_qkv  (f16 out)
  gemm_wmma<true, false><<<(MROWS / 128) * (N3 / 128), 128, 0, stream>>>(
      xh, wqkvp, (void*)qkvh, nullptr, MROWS, N3, DIM_);
  // 4) per-head V transpose
  v_transpose<<<(BB * HH * DHD * NN) / 256, 256, 0, stream>>>(qkvh, vth);
  // 5) flash attention
  attn_flash<<<(BB * HH * (NN / 16)) / 4, 128, 0, stream>>>(qkvh, vth, mask, attnh);
  // 6) out = attn @ w_out + b_out  (fp32 out)
  gemm_wmma<false, true><<<(MROWS / 128) * (DIM_ / 128), 128, 0, stream>>>(
      attnh, woutp, d_out, b_out, MROWS, DIM_, DIM_);
}
